// SympleAgent_29394756173865
// MI455X (gfx1250) — matrix operand order repaired and checked by reading, not compiled, not run on hardware
//
#include <hip/hip_runtime.h>
#include <math.h>

// ---------- types ----------
typedef __attribute__((ext_vector_type(16))) __bf16 v16bf;
typedef __attribute__((ext_vector_type(8)))  __bf16 v8bf;
typedef __attribute__((ext_vector_type(8)))  float  v8f;

#define H      128
#define KTOT   384      // 128 (x) + 256 (h0)
#define NG     512      // 4 gates * 128 (trimmed)
#define NNODES 65535    // internal nodes only (leaves never feed x)

#if defined(__has_builtin)
#if __has_builtin(__builtin_amdgcn_sched_barrier)
#define SCHED_FENCE() __builtin_amdgcn_sched_barrier(0)
#endif
#endif
#ifndef SCHED_FENCE
#define SCHED_FENCE() ((void)0)
#endif

// ---------- helpers ----------
__device__ __forceinline__ unsigned short f2bf(float x) {
    union { float f; unsigned u; } v; v.f = x;
    unsigned r = v.u + 0x7fffu + ((v.u >> 16) & 1u);
    return (unsigned short)(r >> 16);
}
__device__ __forceinline__ float bf2f(unsigned short h) {
    union { unsigned u; float f; } v; v.u = ((unsigned)h) << 16; return v.f;
}
__device__ __forceinline__ float sigm(float x) { return 1.0f / (1.0f + __expf(-x)); }

// A fragment (16x32 bf16): lane<16 -> row=lane, K = kb+{0..15}
//                          lane>=16 -> row=lane-16, K = kb+16+{0..15}
__device__ __forceinline__ v16bf load_a_frag(const unsigned short* p0,
                                             const unsigned short* p1) {
    v8bf lo = *reinterpret_cast<const v8bf*>(p0);
    v8bf hi = *reinterpret_cast<const v8bf*>(p1);
    v16bf a;
#pragma unroll
    for (int i = 0; i < 8; ++i) { a[i] = lo[i]; a[i + 8] = hi[i]; }
    return a;
}

// ---------- prep: pack trimmed weights to bf16, fuse biases ----------
__global__ void prep_weights(const float* __restrict__ W_ih,
                             const float* __restrict__ W_hh,
                             const float* __restrict__ b_ih,
                             const float* __restrict__ b_hh,
                             unsigned short* __restrict__ Weff,
                             float* __restrict__ beff) {
    int n = blockIdx.x;                       // 0..511
    int srow = ((n >> 7) << 8) + (n & 127);   // gate*256 + r : first-H rows of i,f,g,o
    int k = threadIdx.x;                      // 0..383
    float val = (k < 128) ? W_ih[(size_t)srow * 128 + k]
                          : W_hh[(size_t)srow * 256 + (k - 128)];
    Weff[(size_t)n * KTOT + k] = f2bf(val);
    if (k == 0) beff[n] = b_ih[srow] + b_hh[srow];
}

// ---------- prep: gather embeddings (internal nodes only), bf16 ----------
__global__ void prep_emb(const int* __restrict__ node_types,
                         const float* __restrict__ node_args,
                         const float* __restrict__ emb_table,
                         unsigned short* __restrict__ xb) {
    int node = blockIdx.x;                    // 0..NNODES-1
    int j = threadIdx.x;                      // 0..127
    int t = node_types[node];
    float val = emb_table[(size_t)t * 128 + j];
    if (j == 127 && (t == 1 || t == 2)) val = node_args[node];
    xb[(size_t)node * 128 + j] = f2bf(val);
}

// ---------- per-level fused GEMM + LSTM cell (software-pipelined) ----------
// Each wave: one 16x16 output tile for all 4 gates (shared A fragment).
// Block = 256 = 8 waves covering the 128 h-columns. Fragments are
// double-buffered; a sched_barrier pins "issue k+1 loads" before "WMMAs of
// step k" so each WMMA group waits only for its own (older) fragments while
// the 10 newer loads remain in flight.
template <bool FIRST>
__launch_bounds__(256)
__global__ void level_kernel(const unsigned short* __restrict__ xb,
                             const unsigned short* __restrict__ Weff,
                             const float* __restrict__ beff,
                             const unsigned short* __restrict__ h_in,
                             const float* __restrict__ c_in,
                             unsigned short* __restrict__ h_out,
                             float* __restrict__ c_out,
                             int M, int start) {
    const int lane  = threadIdx.x & 31;
    const int wave  = threadIdx.x >> 5;       // 0..7 -> nbase
    const int mbase = blockIdx.x << 4;
    const int nbase = wave << 4;
    const int sel8  = (lane >= 16) ? 8 : 0;   // half-wave selector
    const int rowm  = lane & 15;              // tile row this lane loads (A)
    const int ncol  = nbase + (lane & 15);    // lane's gate-0 W row / out col

    const int NS = FIRST ? 4 : 12;            // k-steps of 32 (x only vs x+h0)

    const unsigned short* Arow_x = xb + (size_t)(start + mbase + rowm) * 128;
    const unsigned short* Arow_h =
        FIRST ? nullptr : (h_in + (size_t)(mbase + rowm) * 256);
    const unsigned short* Bbase[4];
#pragma unroll
    for (int g = 0; g < 4; ++g)
        Bbase[g] = Weff + (size_t)(ncol + g * 128) * KTOT + 2 * sel8;

    v8f acc[4] = {v8f{}, v8f{}, v8f{}, v8f{}};   // i, f, g, o

    // prologue: k-step 0 fragments
    v16bf a_cur = load_a_frag(Arow_x + 2 * sel8, Arow_x + 2 * sel8 + 8);
    v16bf b_cur[4];
#pragma unroll
    for (int g = 0; g < 4; ++g)
        b_cur[g] = *reinterpret_cast<const v16bf*>(Bbase[g]);

#pragma unroll
    for (int ks = 0; ks < NS; ++ks) {
        v16bf a_nxt, b_nxt[4];
        if (ks + 1 < NS) {                    // compile-time after unroll
            const int kn = ks + 1;
            if (kn < 4) {
                const int kb = kn * 32;
                a_nxt = load_a_frag(Arow_x + kb + 2 * sel8, Arow_x + kb + 2 * sel8 + 8);
            } else {
                const int kb = (kn - 4) * 32;
                a_nxt = load_a_frag(Arow_h + kb + 2 * sel8, Arow_h + kb + 2 * sel8 + 8);
            }
#pragma unroll
            for (int g = 0; g < 4; ++g)
                b_nxt[g] = *reinterpret_cast<const v16bf*>(Bbase[g] + kn * 32);
            // keep next-step loads ABOVE this step's WMMAs
            SCHED_FENCE();
        }
#pragma unroll
        for (int g = 0; g < 4; ++g)
            acc[g] = __builtin_amdgcn_wmma_f32_16x16x32_bf16(
                false, a_cur, false, b_cur[g], (short)0, acc[g], false, false);
        if (ks + 1 < NS) {
            a_cur = a_nxt;
#pragma unroll
            for (int g = 0; g < 4; ++g) b_cur[g] = b_nxt[g];
        }
    }

    // ---- fused LSTM cell epilogue ----
    const float bi = beff[ncol];
    const float bf = beff[ncol + 128];
    const float bg = beff[ncol + 256];
    const float bo = beff[ncol + 384];
#pragma unroll
    for (int r = 0; r < 8; ++r) {
        const int m = mbase + r + sel8;       // C/D layout: rows r / r+8
        if (m < M) {
            float iv = acc[0][r] + bi;
            float fv = acc[1][r] + bf;
            float gv = acc[2][r] + bg;
            float ov = acc[3][r] + bo;
            float c0 = FIRST ? 0.0f : c_in[(size_t)(2 * m) * 128 + ncol];
            float cn = sigm(fv) * c0 + sigm(iv) * tanhf(gv);
            float hn = sigm(ov) * tanhf(cn);
            c_out[(size_t)m * 128 + ncol] = cn;
            h_out[(size_t)m * 128 + ncol] = f2bf(hn);
        }
    }
}

// ---------- root MLP + masked softmax (tiny, scalar) ----------
__global__ void final_kernel(const unsigned short* __restrict__ hroot,
                             const float* __restrict__ W1, const float* __restrict__ b1,
                             const float* __restrict__ W2, const float* __restrict__ b2,
                             const float* __restrict__ vmask,
                             float* __restrict__ out) {
    __shared__ float hs[128], as[128], ls[32], es[32];
    int t = threadIdx.x;                      // 0..127
    hs[t] = bf2f(hroot[t]);
    __syncthreads();
    float acc = b1[t];
#pragma unroll 8
    for (int k = 0; k < 128; ++k) acc += hs[k] * W1[(size_t)t * 128 + k];
    as[t] = fmaxf(acc, 0.0f);
    __syncthreads();
    if (t < 32) {
        float l = b2[t];
#pragma unroll 8
        for (int k = 0; k < 128; ++k) l += as[k] * W2[(size_t)t * 128 + k];
        l += __logf(vmask[t]);
        ls[t] = l * (1.0f / 3.0f);            // TEMPERATURE
    }
    __syncthreads();
    if (t < 32) {
        float m = -1e30f;
        for (int k = 0; k < 32; ++k) m = fmaxf(m, ls[k]);
        es[t] = __expf(ls[t] - m);
    }
    __syncthreads();
    if (t < 32) {
        float s = 0.0f;
        for (int k = 0; k < 32; ++k) s += es[k];
        out[t] = es[t] / s;
    }
}

// ---------- host launch ----------
extern "C" void kernel_launch(void* const* d_in, const int* in_sizes, int n_in,
                              void* d_out, int out_size, void* d_ws, size_t ws_size,
                              hipStream_t stream) {
    const int*   node_types = (const int*)  d_in[0];
    const float* node_args  = (const float*)d_in[1];
    const float* vmask      = (const float*)d_in[2];
    const float* emb_table  = (const float*)d_in[3];
    const float* W_ih       = (const float*)d_in[4];
    const float* W_hh       = (const float*)d_in[5];
    const float* b_ih       = (const float*)d_in[6];
    const float* b_hh       = (const float*)d_in[7];
    const float* W1         = (const float*)d_in[8];
    const float* b1         = (const float*)d_in[9];
    const float* W2         = (const float*)d_in[10];
    const float* b2         = (const float*)d_in[11];

    char* ws = (char*)d_ws;
    // workspace layout (offsets 256B aligned)
    unsigned short* Weff = (unsigned short*)(ws + 0);             // 512*384*2  = 393216
    float*          beff = (float*)         (ws + 393216);        // 512*4      = 2048
    unsigned short* xb   = (unsigned short*)(ws + 395264);        // 65535*128*2= 16776960
    unsigned short* hA   = (unsigned short*)(ws + 17172224);      // 32768*128*2= 8388608
    unsigned short* hB   = (unsigned short*)(ws + 25560832);      // 8388608
    float*          cA   = (float*)         (ws + 33949440);      // 32768*128*4= 16777216
    float*          cB   = (float*)         (ws + 50726656);      // 16777216 -> ~67.5MB

    prep_weights<<<NG, KTOT, 0, stream>>>(W_ih, W_hh, b_ih, b_hh, Weff, beff);
    prep_emb<<<NNODES, 128, 0, stream>>>(node_types, node_args, emb_table, xb);

    unsigned short *h_in = nullptr, *h_out = hA;
    float          *c_in = nullptr, *c_out = cA;
    for (int lvl = 15; lvl >= 0; --lvl) {
        const int M = 1 << lvl;
        const int start = M - 1;
        const int grid = (M + 15) / 16;
        if (lvl == 15) {
            level_kernel<true><<<grid, 256, 0, stream>>>(
                xb, Weff, beff, h_in, c_in, h_out, c_out, M, start);
        } else {
            level_kernel<false><<<grid, 256, 0, stream>>>(
                xb, Weff, beff, h_in, c_in, h_out, c_out, M, start);
        }
        h_in = h_out; c_in = c_out;
        h_out = (h_out == hA) ? hB : hA;
        c_out = (c_out == cA) ? cB : cA;
    }

    final_kernel<<<1, 128, 0, stream>>>(h_in, W1, b1, W2, b2, vmask, (float*)d_out);
}
